// TransformerAutoencoder_40707700032086
// MI455X (gfx1250) — matrix operand order repaired and checked by reading, not compiled, not run on hardware
//
#include <hip/hip_runtime.h>
#include <hip/hip_bf16.h>

// ---------------------------------------------------------------------------
// Types for CDNA5 WMMA
// ---------------------------------------------------------------------------
typedef __attribute__((ext_vector_type(16))) __bf16          v16bf;
typedef __attribute__((ext_vector_type(8)))  float           v8f;
typedef __attribute__((ext_vector_type(16))) unsigned short  v16us;

__device__ __forceinline__ unsigned short f2bf(float f) {
    unsigned u = __float_as_uint(f);
    // round-to-nearest-even to bf16
    unsigned r = (u + 0x7FFFu + ((u >> 16) & 1u)) >> 16;
    return (unsigned short)r;
}

// Build one 16x32 bf16 A fragment (contiguous K order, elem e <-> K = kSel + e) from a row-major
// f32 row pointer already adjusted for k0 + lane K-select.
__device__ __forceinline__ v16bf make_afrag(const float* p) {
    const float4* c0 = (const float4*)(p);
    const float4* c1 = (const float4*)(p + 8);
    float4 f0 = c0[0], f1 = c0[1], f2 = c1[0], f3 = c1[1];
    float af[16] = { f0.x, f0.y, f0.z, f0.w, f1.x, f1.y, f1.z, f1.w,
                     f2.x, f2.y, f2.z, f2.w, f3.x, f3.y, f3.z, f3.w };
    v16us au;
    #pragma unroll
    for (int e = 0; e < 16; e++) au[e] = f2bf(af[e]);
    return __builtin_bit_cast(v16bf, au);
}

// Async global->LDS copy of 32 bytes per lane (two b128 ops; INST_OFFSET
// applies to both the LDS and the global address). ASYNCcnt-tracked.
__device__ __forceinline__ void async_copy32(unsigned ldsOff, unsigned long long gaddr) {
    asm volatile(
        "global_load_async_to_lds_b128 %0, %1, off\n\t"
        "global_load_async_to_lds_b128 %0, %1, off offset:16"
        :: "v"(ldsOff), "v"(gaddr) : "memory");
}

// ---------------------------------------------------------------------------
// Repack a row-major f32 weight W[K][N] into WMMA B-matrix tile order (bf16).
// Tile = 32(K) x 16(N). Per-lane layout (wave32, 16-bit B operand):
//   lanes 0-15 hold K=0..15 (elem e <-> K=e), lanes 16-31 hold K=16..31.
//   N = lane % 16.  Packed: tile*(32*16) + lane*16 + e   (contiguous per lane)
// Tiles ordered kt-major, nt-minor => one K-step's N-panel is contiguous.
// ---------------------------------------------------------------------------
__global__ __launch_bounds__(256) void repack_w_kernel(
    const float* __restrict__ W, unsigned short* __restrict__ out, int K, int N)
{
    long gid = (long)blockIdx.x * 256 + threadIdx.x;
    long total = (long)K * N;
    if (gid >= total) return;
    int e    = (int)(gid & 15);
    int l    = (int)((gid >> 4) & 31);
    long tile = gid >> 9;
    int ntN  = N >> 4;
    int nt   = (int)(tile % ntN);
    int kt   = (int)(tile / ntN);
    int k    = kt * 32 + ((l & 16) ? 16 : 0) + e;
    int n    = nt * 16 + (l & 15);
    out[gid] = f2bf(W[(long)k * N + n]);
}

// ---------------------------------------------------------------------------
// GEMM: C[M,N] = act(A[M,K] @ W[K,N] + bias). A row-major f32, W pre-packed
// bf16. Block = 8 waves = 256(M) x 128(N). Shared 32(K)x128(N) bf16 B panel
// staged in LDS via global_load_async_to_lds_b128, double-buffered with a
// single barrier per K-step (the barrier that publishes panel i also proves
// panel i-1 is fully consumed, so the next async issue is race-free and the
// copy overlaps the whole WMMA chain). Each wave: 32x128 strip =
// 16 x v_wmma_f32_16x16x32_bf16 per K-step; each LDS B fragment feeds 2 WMMAs.
// ---------------------------------------------------------------------------
__global__ __launch_bounds__(256) void gemm_bf16_kernel(
    const float* __restrict__ A, const unsigned short* __restrict__ Wp,
    const float* __restrict__ bias, float* __restrict__ C,
    int M, int N, int K, int ldc, int relu)
{
    __shared__ __align__(64) unsigned short smem[2][4096];  // 2 x 8KB B panels

    const int nStrips = N >> 7;
    const int mB   = (int)blockIdx.x / nStrips;
    const int ns   = (int)blockIdx.x % nStrips;
    const int nBase = ns << 7;
    const int tid  = threadIdx.x;
    const int wid  = tid >> 5;
    const int lane = tid & 31;

    const int row0 = (mB << 8) + (wid << 5) + (lane & 15);   // M half 0
    const int kSel = (lane & 16) ? 16 : 0;     // A: high lanes take K+16..K+31
    const float* arow0 = A + (long)row0 * K;
    const float* arow1 = arow0 + (long)16 * K;               // M half 1

    // global source for this block's B panels: contiguous 8KB per K-step
    const int ntN = N >> 4;
    unsigned long long g =
        (unsigned long long)(const char*)Wp +
        ((unsigned long long)(nBase >> 4)) * 1024ull + (unsigned)tid * 32u;
    const unsigned long long gstep = (unsigned long long)ntN * 1024ull;

    unsigned ldsBase0 = (unsigned)(size_t)(&smem[0][0]) + (unsigned)tid * 32u;
    unsigned ldsBase1 = (unsigned)(size_t)(&smem[1][0]) + (unsigned)tid * 32u;

    v8f acc[16];
    #pragma unroll
    for (int t = 0; t < 16; t++) acc[t] = (v8f)(0.0f);

    const int nIter = K >> 5;

    // prologue: stage panel 0
    async_copy32(ldsBase0, g);
    g += gstep;

    for (int i = 0; i < nIter; ++i) {
        const int cur = i & 1;

        // panel i landed (completion is in-order; only panel i outstanding)
        asm volatile("s_wait_asynccnt 0x0" ::: "memory");
        __syncthreads();  // publishes panel i AND proves panel i-1 consumed

        // overlap: stage panel i+1 into the buffer just freed
        if (i + 1 < nIter) {
            async_copy32(cur ? ldsBase0 : ldsBase1, g);
            g += gstep;
        }

        // ---- two A fragments (M halves), f32 -> bf16 RNE ----
        const int k0 = i << 5;
        v16bf a0 = make_afrag(arow0 + k0 + kSel);
        v16bf a1 = make_afrag(arow1 + k0 + kSel);

        // ---- 8 B fragments from LDS; each feeds both M halves ----
        const unsigned short* bpanel = &smem[cur][0];
        #pragma unroll
        for (int t = 0; t < 8; t++) {
            v16us bu = *(const v16us*)(bpanel + t * 512 + lane * 16);
            v16bf bfrag = __builtin_bit_cast(v16bf, bu);
            acc[t]     = __builtin_amdgcn_wmma_f32_16x16x32_bf16(
                             false, a0, false, bfrag, (short)0, acc[t],     false, false);
            acc[8 + t] = __builtin_amdgcn_wmma_f32_16x16x32_bf16(
                             false, a1, false, bfrag, (short)0, acc[8 + t], false, false);
        }
    }

    // ---- store: C/D layout: VGPR r -> M = r (lanes 0-15) or 8+r (16-31) ----
    const int rbase = (lane & 16) ? 8 : 0;
    const int col0  = nBase + (lane & 15);
    #pragma unroll
    for (int h = 0; h < 2; h++) {
        const int mrow0 = (mB << 8) + (wid << 5) + h * 16 + rbase;
        #pragma unroll
        for (int t = 0; t < 8; t++) {
            int col = col0 + t * 16;
            float bs = bias ? bias[col] : 0.0f;
            #pragma unroll
            for (int r = 0; r < 8; r++) {
                float v = acc[h * 8 + t][r] + bs;
                if (relu) v = fmaxf(v, 0.0f);
                C[(long)(mrow0 + r) * ldc + col] = v;
            }
        }
    }
}

// ---------------------------------------------------------------------------
// BatchNorm stats over (B, A) per channel (6 channels), x layout [B, A, 6]
// stats[c] = mean, stats[6+c] = rsqrt(var + eps)
// ---------------------------------------------------------------------------
__global__ __launch_bounds__(256) void bn_stats_kernel(
    const float* __restrict__ x, float* __restrict__ stats)
{
    int c = blockIdx.x;
    __shared__ float s1[256], s2[256];
    float a = 0.f, b = 0.f;
    for (int i = threadIdx.x; i < 2048 * 64; i += 256) {
        float v = x[(long)i * 6 + c];
        a += v; b += v * v;
    }
    s1[threadIdx.x] = a; s2[threadIdx.x] = b;
    __syncthreads();
    for (int st = 128; st; st >>= 1) {
        if ((int)threadIdx.x < st) {
            s1[threadIdx.x] += s1[threadIdx.x + st];
            s2[threadIdx.x] += s2[threadIdx.x + st];
        }
        __syncthreads();
    }
    if (threadIdx.x == 0) {
        const float inv = 1.0f / 131072.0f;
        float mu  = s1[0] * inv;
        float var = s2[0] * inv - mu * mu;
        stats[c]     = mu;
        stats[6 + c] = rsqrtf(var + 1e-5f);
    }
}

// ---------------------------------------------------------------------------
// BN-normalize + Conv1d(6->16, K=5, pad=2) + flatten -> xf[B, 16*64]
// ---------------------------------------------------------------------------
__global__ __launch_bounds__(256) void bnconv_kernel(
    const float* __restrict__ x, const float* __restrict__ stats,
    const float* __restrict__ bn_g, const float* __restrict__ bn_b,
    const float* __restrict__ conv_w, const float* __restrict__ conv_b,
    float* __restrict__ xf)
{
    int gid = blockIdx.x * 256 + threadIdx.x;  // 2048*16*64 total
    int a  = gid & 63;
    int cc = (gid >> 6) & 15;
    int b  = gid >> 10;
    float sum = conv_b[cc];
    #pragma unroll
    for (int ci = 0; ci < 6; ci++) {
        float gain  = stats[6 + ci] * bn_g[ci];
        float shift = bn_b[ci] - stats[ci] * gain;
        #pragma unroll
        for (int k = 0; k < 5; k++) {
            int p = a + k - 2;
            if (p >= 0 && p < 64) {
                float v = x[((long)b * 64 + p) * 6 + ci] * gain + shift;
                sum += v * conv_w[(cc * 6 + ci) * 5 + k];
            }
        }
    }
    xf[gid] = sum;
}

// ---------------------------------------------------------------------------
// LayerNorm over last dim D=128; one wave32 per row (4 f32 per lane)
// ---------------------------------------------------------------------------
__global__ __launch_bounds__(256) void layernorm_kernel(
    const float* __restrict__ X, const float* __restrict__ g,
    const float* __restrict__ b, float* __restrict__ Y, int M)
{
    int wid = threadIdx.x >> 5, lane = threadIdx.x & 31;
    int row = blockIdx.x * 8 + wid;
    if (row >= M) return;
    const float* x = X + (long)row * 128;
    float4 v = *(const float4*)(x + lane * 4);
    float s = v.x + v.y + v.z + v.w;
    #pragma unroll
    for (int m = 16; m; m >>= 1) s += __shfl_xor(s, m, 32);
    float mu = s * (1.0f / 128.0f);
    float dx = v.x - mu, dy = v.y - mu, dz = v.z - mu, dw = v.w - mu;
    float q = dx * dx + dy * dy + dz * dz + dw * dw;
    #pragma unroll
    for (int m = 16; m; m >>= 1) q += __shfl_xor(q, m, 32);
    float rstd = rsqrtf(q * (1.0f / 128.0f) + 1e-5f);
    float4 gg = *(const float4*)(g + lane * 4);
    float4 bb = *(const float4*)(b + lane * 4);
    float4 o;
    o.x = dx * rstd * gg.x + bb.x;
    o.y = dy * rstd * gg.y + bb.y;
    o.z = dz * rstd * gg.z + bb.z;
    o.w = dw * rstd * gg.w + bb.w;
    *(float4*)(Y + (long)row * 128 + lane * 4) = o;
}

// ---------------------------------------------------------------------------
// Scatter token projections [B*16, 128] -> cache[B, 32, 128] rows 0..15
// ---------------------------------------------------------------------------
__global__ __launch_bounds__(256) void scatter_tokens_kernel(
    const float* __restrict__ in, float* __restrict__ cache)
{
    long gid = (long)blockIdx.x * 256 + threadIdx.x;  // 32768*128
    int d = (int)(gid & 127);
    long r = gid >> 7;
    long bt = r >> 4;
    int  tt = (int)(r & 15);
    cache[bt * 4096 + (long)tt * 128 + d] = in[gid];
}

// ---------------------------------------------------------------------------
// One attention step: scores over valid cache rows, softmax, weighted V + pos
// ---------------------------------------------------------------------------
__global__ __launch_bounds__(128) void attn_step_kernel(
    const float* __restrict__ kc, const float* __restrict__ vc,
    const float* __restrict__ pos, float* __restrict__ att, int step)
{
    int b = blockIdx.x, t = threadIdx.x;
    __shared__ float ps[128];
    __shared__ float sw[32];
    ps[t] = pos[step * 128 + t];
    __syncthreads();
    int lim = 16 + step;
    if (t < 32) {
        float s = 0.0f;
        if (t < lim) {
            const float* kr = kc + (long)b * 4096 + t * 128;
            #pragma unroll 4
            for (int d = 0; d < 128; d++) s += kr[d] * ps[d];
            s *= 0.08838834764831845f;  // 1/sqrt(128)
        }
        sw[t] = s;
    }
    __syncthreads();
    if (t == 0) {
        float mx = -1e30f;
        for (int s = 0; s < lim; s++) mx = fmaxf(mx, sw[s]);
        float Z = 0.0f;
        for (int s = 0; s < lim; s++) { float e = __expf(sw[s] - mx); sw[s] = e; Z += e; }
        float inv = 1.0f / Z;
        for (int s = 0; s < lim; s++) sw[s] *= inv;
    }
    __syncthreads();
    float acc = ps[t];
    const float* vb = vc + (long)b * 4096;
    for (int s = 0; s < lim; s++) acc += sw[s] * vb[s * 128 + t];
    att[(long)b * 128 + t] = acc;
}

// ---------------------------------------------------------------------------
// Softmax over E=1024 logits, CDF inverse-sample, gather embedding column.
// Straight-through estimator is identity in forward: tok == emb.T[idx].
// ---------------------------------------------------------------------------
__global__ __launch_bounds__(256) void sample_step_kernel(
    const float* __restrict__ logits, const float* __restrict__ u,
    const float* __restrict__ emb, float* __restrict__ out_tokens, int step)
{
    int b = blockIdx.x, t = threadIdx.x;
    __shared__ float es[1024];
    __shared__ float red[256];
    __shared__ int   sidx;
    const float* lr = logits + (long)b * 1024;
    float mx = -1e30f;
    for (int j = t; j < 1024; j += 256) { float v = lr[j]; es[j] = v; mx = fmaxf(mx, v); }
    red[t] = mx; __syncthreads();
    for (int st = 128; st; st >>= 1) {
        if (t < st) red[t] = fmaxf(red[t], red[t + st]);
        __syncthreads();
    }
    mx = red[0]; __syncthreads();
    float ls = 0.0f;
    for (int j = t; j < 1024; j += 256) { float e = __expf(es[j] - mx); es[j] = e; ls += e; }
    red[t] = ls; __syncthreads();
    for (int st = 128; st; st >>= 1) {
        if (t < st) red[t] += red[t + st];
        __syncthreads();
    }
    if (t == 0) {
        float Z = red[0];
        float target = u[(long)step * 2048 + b] * Z;  // cdf<u  <=>  prefix < u*Z
        float c = 0.0f; int idx = 0;
        for (int j = 0; j < 1024; j++) { c += es[j]; if (c < target) idx++; }
        if (idx > 1023) idx = 1023;
        sidx = idx;
    }
    __syncthreads();
    if (t < 128) out_tokens[(long)b * 2048 + step * 128 + t] = emb[(long)t * 1024 + sidx];
}

// ---------------------------------------------------------------------------
// Host orchestration
// ---------------------------------------------------------------------------
extern "C" void kernel_launch(void* const* d_in, const int* in_sizes, int n_in,
                              void* d_out, int out_size, void* d_ws, size_t ws_size,
                              hipStream_t stream)
{
    const float* x       = (const float*)d_in[0];
    const float* u       = (const float*)d_in[1];
    const float* bn_g    = (const float*)d_in[2];
    const float* bn_b    = (const float*)d_in[3];
    const float* conv_w  = (const float*)d_in[4];
    const float* conv_b  = (const float*)d_in[5];
    const float* tok_w1  = (const float*)d_in[6];
    const float* tok_b1  = (const float*)d_in[7];
    const float* tok_w2  = (const float*)d_in[8];
    const float* tok_b2  = (const float*)d_in[9];
    const float* pln_g   = (const float*)d_in[10];
    const float* pln_b   = (const float*)d_in[11];
    const float* proj_w  = (const float*)d_in[12];
    const float* proj_b  = (const float*)d_in[13];
    const float* emb     = (const float*)d_in[14];
    const float* pos     = (const float*)d_in[15];
    const float* mlp_w1  = (const float*)d_in[16];
    const float* mlp_b1  = (const float*)d_in[17];
    const float* mlp_w2  = (const float*)d_in[18];
    const float* mlp_b2  = (const float*)d_in[19];
    const float* d1w1    = (const float*)d_in[20];
    const float* d1b1    = (const float*)d_in[21];
    const float* d1w2    = (const float*)d_in[22];
    const float* d1b2    = (const float*)d_in[23];
    const float* d2w1    = (const float*)d_in[24];
    const float* d2b1    = (const float*)d_in[25];
    const float* d2w2    = (const float*)d_in[26];
    const float* d2b2    = (const float*)d_in[27];
    const float* d3w1    = (const float*)d_in[28];
    const float* d3b1    = (const float*)d_in[29];
    const float* d3w2    = (const float*)d_in[30];
    const float* d3b2    = (const float*)d_in[31];

    const int B = 2048;
    const size_t MB = 1024 * 1024;

    // ---- workspace bump allocator ----
    size_t off = 0;
    auto balloc = [&](size_t bytes) -> void* {
        void* p = (char*)d_ws + off;
        off = (off + bytes + 255) & ~(size_t)255;
        return p;
    };

    float* stats = (float*)balloc(64);

    unsigned short* pw_tok1 = (unsigned short*)balloc((size_t)1024 * 2048 * 2);
    unsigned short* pw_tok2 = (unsigned short*)balloc((size_t)2048 * 2048 * 2);
    unsigned short* pw_proj = (unsigned short*)balloc((size_t)4 * 128 * 128 * 2);
    unsigned short* pw_mlp1 = (unsigned short*)balloc((size_t)128 * 256 * 2);
    unsigned short* pw_mlp2 = (unsigned short*)balloc((size_t)256 * 128 * 2);
    unsigned short* pw_emb  = (unsigned short*)balloc((size_t)128 * 1024 * 2);
    unsigned short* pw_d1a  = (unsigned short*)balloc((size_t)2048 * 4096 * 2);
    unsigned short* pw_d1b  = (unsigned short*)balloc((size_t)4096 * 2048 * 2);
    unsigned short* pw_d2a  = (unsigned short*)balloc((size_t)2048 * 4096 * 2);
    unsigned short* pw_d2b  = (unsigned short*)balloc((size_t)4096 * 2048 * 2);
    unsigned short* pw_d3a  = (unsigned short*)balloc((size_t)2048 * 4096 * 2);
    unsigned short* pw_d3b  = (unsigned short*)balloc((size_t)4096 * 384 * 2);

    float* xf     = (float*)balloc(8 * MB);   // [2048,1024]; reused as logits [2048,1024]
    float* bufA   = (float*)balloc(16 * MB);  // tok hidden / proj tmp / dec1 out
    float* bufB   = (float*)balloc(16 * MB);  // tokens / out_tokens
    float* bufC   = (float*)balloc(16 * MB);  // token LN / dec2 out
    float* bufD   = (float*)balloc(32 * MB);  // 4096-wide hiddens
    float* kc     = (float*)balloc(32 * MB);  // [2048,32,128]
    float* vc     = (float*)balloc(32 * MB);
    float* att    = (float*)balloc(1 * MB);   // [2048,128]
    float* lna    = (float*)balloc(1 * MB);
    float* mlph   = (float*)balloc(2 * MB);   // [2048,256]
    float* mlptok = (float*)balloc(1 * MB);   // [2048,128]
    float* logits = xf;
    (void)ws_size; (void)n_in; (void)in_sizes; (void)out_size;

    auto repack = [&](const float* W, unsigned short* outp, int K, int N) {
        long total = (long)K * N;
        repack_w_kernel<<<dim3((unsigned)((total + 255) / 256)), dim3(256), 0, stream>>>(W, outp, K, N);
    };
    auto gemm = [&](const float* A, const unsigned short* Wp, const float* bias,
                    float* C, int M, int N, int K, int ldc, int relu) {
        int blocks = (M >> 8) * (N >> 7);
        gemm_bf16_kernel<<<dim3((unsigned)blocks), dim3(256), 0, stream>>>(
            A, Wp, bias, C, M, N, K, ldc, relu);
    };

    // ---- pack all weights to WMMA bf16 tile order ----
    repack(tok_w1, pw_tok1, 1024, 2048);
    repack(tok_w2, pw_tok2, 2048, 2048);
    for (int i = 0; i < 4; i++)
        repack(proj_w + (size_t)i * 128 * 128, pw_proj + (size_t)i * 128 * 128, 128, 128);
    repack(mlp_w1, pw_mlp1, 128, 256);
    repack(mlp_w2, pw_mlp2, 256, 128);
    repack(emb,    pw_emb,  128, 1024);
    repack(d1w1, pw_d1a, 2048, 4096);
    repack(d1w2, pw_d1b, 4096, 2048);
    repack(d2w1, pw_d2a, 2048, 4096);
    repack(d2w2, pw_d2b, 4096, 2048);
    repack(d3w1, pw_d3a, 2048, 4096);
    repack(d3w2, pw_d3b, 4096, 384);

    // ---- BN stats + conv + flatten ----
    bn_stats_kernel<<<dim3(6), dim3(256), 0, stream>>>(x, stats);
    bnconv_kernel<<<dim3(8192), dim3(256), 0, stream>>>(x, stats, bn_g, bn_b, conv_w, conv_b, xf);

    // ---- tokenizer MLP: [2048,1024] -> relu [2048,2048] -> [2048,2048] ----
    gemm(xf,   pw_tok1, tok_b1, bufA, B, 2048, 1024, 2048, 1);
    gemm(bufA, pw_tok2, tok_b2, bufB, B, 2048, 2048, 2048, 0);  // tokens [B,16,128]

    // ---- KV caches: zero, then project input tokens into rows 0..15 ----
    hipMemsetAsync(kc, 0, 32 * MB, stream);
    hipMemsetAsync(vc, 0, 32 * MB, stream);
    for (int i = 0; i < 2; i++) {
        layernorm_kernel<<<dim3(4096), dim3(256), 0, stream>>>(
            bufB, pln_g + (size_t)i * 128, pln_b + (size_t)i * 128, bufC, B * 16);
        gemm(bufC, pw_proj + (size_t)i * 128 * 128, proj_b + (size_t)i * 128,
             bufA, B * 16, 128, 128, 128, 0);
        scatter_tokens_kernel<<<dim3((B * 16 * 128) / 256), dim3(256), 0, stream>>>(
            bufA, (i == 0) ? kc : vc);
    }

    // ---- autoregressive decode: 16 steps ----
    float* out_tokens = bufB;  // tokens dead; reuse as [B, 16*128]
    for (int i = 0; i < 16; i++) {
        attn_step_kernel<<<dim3(B), dim3(128), 0, stream>>>(kc, vc, pos, att, i);

        // k_cache[:, 16+i] = LN2(att) @ Wk ; v_cache[:, 16+i] = LN3(att) @ Wv
        layernorm_kernel<<<dim3(256), dim3(256), 0, stream>>>(
            att, pln_g + 2 * 128, pln_b + 2 * 128, lna, B);
        gemm(lna, pw_proj + (size_t)2 * 128 * 128, proj_b + 2 * 128,
             kc + (size_t)(16 + i) * 128, B, 128, 128, 4096, 0);
        layernorm_kernel<<<dim3(256), dim3(256), 0, stream>>>(
            att, pln_g + 3 * 128, pln_b + 3 * 128, lna, B);
        gemm(lna, pw_proj + (size_t)3 * 128 * 128, proj_b + 3 * 128,
             vc + (size_t)(16 + i) * 128, B, 128, 128, 4096, 0);

        // token MLP + logits + sampling gather
        gemm(att,    pw_mlp1, mlp_b1, mlph,   B, 256, 128, 256, 1);
        gemm(mlph,   pw_mlp2, mlp_b2, mlptok, B, 128, 256, 128, 0);
        gemm(mlptok, pw_emb,  nullptr, logits, B, 1024, 128, 1024, 0);
        sample_step_kernel<<<dim3(B), dim3(256), 0, stream>>>(
            logits, u, emb, out_tokens, i);
    }

    // ---- decoder MLP stack ----
    gemm(out_tokens, pw_d1a, d1b1, bufD, B, 4096, 2048, 4096, 1);
    gemm(bufD,       pw_d1b, d1b2, bufA, B, 2048, 4096, 2048, 0);
    gemm(bufA,       pw_d2a, d2b1, bufD, B, 4096, 2048, 4096, 1);
    gemm(bufD,       pw_d2b, d2b2, bufC, B, 2048, 4096, 2048, 0);
    gemm(bufC,       pw_d3a, d3b1, bufD, B, 4096, 2048, 4096, 1);
    gemm(bufD,       pw_d3b, d3b2, (float*)d_out, B, 384, 4096, 384, 0);
}
